// LMU_45483703665332
// MI455X (gfx1250) — matrix operand order repaired and checked by reading, not compiled, not run on hardware
//
#include <hip/hip_runtime.h>
#include <hip/hip_bf16.h>
#include <cstdint>
#include <cstddef>

#define Dm    768
#define Lseq  128
#define Bsz   256
#define PADN  784
#define NSQ   (PADN * PADN)
#define THETA 128.0f
#define EXP_S 14
#define TAYLOR_K 12

typedef __attribute__((ext_vector_type(16))) __bf16 v16bf;
typedef __attribute__((ext_vector_type(8)))  float  v8f;
typedef __attribute__((ext_vector_type(2)))  float  v2f;
typedef __attribute__((ext_vector_type(4)))  unsigned int u32x4;
typedef __attribute__((ext_vector_type(8)))  int i32x8;
typedef __attribute__((ext_vector_type(4)))  int i32x4;

__device__ __forceinline__ unsigned short f2bf(float f) {
  union { float f; uint32_t u; } c; c.f = f;
  return (unsigned short)((c.u + 0x7FFFu + ((c.u >> 16) & 1u)) >> 16);
}

union FragB16 { uint4 q[2]; v16bf v; };

#if __has_builtin(__builtin_amdgcn_tensor_load_to_lds) && \
    __has_builtin(__builtin_amdgcn_s_wait_tensorcnt)
#define HAVE_TDM 1
#else
#define HAVE_TDM 0
#endif

#if HAVE_TDM
// ---------------------------------------------------------------------------
// TDM: DMA a 2D tile (width_u32 DWORDs x rows) from global into LDS, adding
// 4 DWORDs of padding after every 16 DWORDs (-> 80-byte LDS row stride, which
// matches the [128][40]-ushort LDS tiles). D# bit layout per ISA ch.8.
// Issue from one wave; completion via TENSORcnt.
// ---------------------------------------------------------------------------
__device__ __forceinline__ void tdm_load_2d(const void* gsrc, unsigned lds_addr,
                                            unsigned width_u32, unsigned rows,
                                            unsigned row_stride_u32) {
  unsigned long long ga = (unsigned long long)(uintptr_t)gsrc;
  u32x4 g0;
  g0[0] = 1u;                                    // count=1 (valid user D#)
  g0[1] = lds_addr;                              // LDS byte address
  g0[2] = (unsigned)(ga & 0xFFFFFFFFu);          // global_addr[31:0]
  g0[3] = (unsigned)((ga >> 32) & 0x01FFFFFFu)   // global_addr[56:32]
          | 0x80000000u;                         // type = 2 ("image")
  i32x8 g1;
  g1[0] = (int)((2u << 16)                       // data_size: 4-byte units
                | (1u << 20)                     // pad_enable
                | (3u << 22)                     // pad_interval: 16 DWORDs
                | (3u << 25));                   // pad_amount: 4 DWORDs
  g1[1] = (int)((width_u32 & 0xFFFFu) << 16);    // tensor_dim0[15:0]
  g1[2] = (int)(((width_u32 >> 16) & 0xFFFFu)    // tensor_dim0[31:16]
                | ((rows & 0xFFFFu) << 16));     // tensor_dim1[15:0]
  g1[3] = (int)(((rows >> 16) & 0xFFFFu)         // tensor_dim1[31:16]
                | ((width_u32 & 0xFFFFu) << 16));// tile_dim0
  g1[4] = (int)(rows & 0xFFFFu);                 // tile_dim1 (tile_dim2 = 0)
  g1[5] = (int)row_stride_u32;                   // tensor_dim0_stride[31:0]
  g1[6] = 0;                                     // stride0 hi / stride1 lo
  g1[7] = 0;
  i32x4 gz4 = {0, 0, 0, 0};
  i32x8 gz8 = {0, 0, 0, 0, 0, 0, 0, 0};
  __builtin_amdgcn_tensor_load_to_lds(g0, g1, gz4, gz4, gz8, 0);
}
#endif

// ---------------------------------------------------------------------------
// expm setup: X = Maug / 2^EXP_S, zero-padded 784x784
// ---------------------------------------------------------------------------
__global__ void k_build_X(float* __restrict__ X) {
  int idx = blockIdx.x * 256 + threadIdx.x;
  if (idx >= NSQ) return;
  int i = idx / PADN, j = idx % PADN;
  float v = 0.f;
  const float sc = 1.0f / (float)(1 << EXP_S);
  if (i < Dm) {
    float R = (2.0f * i + 1.0f) / THETA;
    if (j < Dm) {
      float a = (i < j) ? -1.0f : (((i - j + 1) & 1) ? -1.0f : 1.0f);
      v = a * R * sc;
    } else if (j == Dm) {
      v = ((i & 1) ? -R : R) * sc;   // (-1)^i * R
    }
  }
  X[idx] = v;
}

__global__ void k_init_EP(const float* __restrict__ X, float* __restrict__ E,
                          float* __restrict__ P) {
  int idx = blockIdx.x * 256 + threadIdx.x;
  if (idx >= NSQ) return;
  int i = idx / PADN, j = idx % PADN;
  float xv = X[idx];
  P[idx] = xv;
  E[idx] = xv + (i == j ? 1.0f : 0.0f);
}

__global__ void k_axpy(float* __restrict__ E, const float* __restrict__ P, int n) {
  int idx = blockIdx.x * 256 + threadIdx.x;
  if (idx < n) E[idx] += P[idx];
}

// ---------------------------------------------------------------------------
// Generic fp32 WMMA GEMM: C = alpha * A @ op(B). Row-major. Edge handling is
// fully branchless: loads use clamped (always-valid) addresses and the result
// is multiplied by a 0/1 mask, so the value is unconditionally consumed and
// the compiler cannot predicate the load (no exec-mask branches in staging).
// Block 256 thr = 8 waves (4 M-groups x 2 N-groups), tile 64x64, BK=16.
// Uses V_WMMA_F32_16X16X4_F32.
// ---------------------------------------------------------------------------
__global__ __launch_bounds__(256) void k_gemm_f32(
    const float* __restrict__ A, int lda, const float* __restrict__ B, int ldb,
    float* __restrict__ C, int ldc, int M, int N, int K, float alpha, int transB) {
  __shared__ __align__(16) float As[64][20];
  __shared__ __align__(16) float Bs[64][20];   // Bs[n][k]
  const int t = threadIdx.x;
  const int m0 = blockIdx.y * 64, n0 = blockIdx.x * 64;
  const int wave = t >> 5, lane = t & 31;
  const int wm = wave & 3, wn = wave >> 2;
  const int lm = lane & 15;
  const int kh = (lane >> 4) << 1;   // K offset 0 or 2 within a K=4 step
  v8f acc[2];
#pragma unroll
  for (int j = 0; j < 2; ++j)
#pragma unroll
    for (int e = 0; e < 8; ++e) acc[j][e] = 0.f;

  for (int k0 = 0; k0 < K; k0 += 16) {
#pragma unroll
    for (int i = 0; i < 4; ++i) {
      int e = i * 256 + t;
      int r = e >> 4, kk = e & 15;
      int gr = m0 + r, gk = k0 + kk;
      int rc = gr < M ? gr : M - 1;
      int kc = gk < K ? gk : K - 1;
      float msk = (gr < M && gk < K) ? 1.f : 0.f;
      As[r][kk] = A[(size_t)rc * lda + kc] * msk;
    }
    if (transB) {
#pragma unroll
      for (int i = 0; i < 4; ++i) {
        int e = i * 256 + t;
        int n = e >> 4, kk = e & 15;
        int gn = n0 + n, gk = k0 + kk;
        int nc = gn < N ? gn : N - 1;
        int kc = gk < K ? gk : K - 1;
        float msk = (gn < N && gk < K) ? 1.f : 0.f;
        Bs[n][kk] = B[(size_t)nc * ldb + kc] * msk;
      }
    } else {
#pragma unroll
      for (int i = 0; i < 4; ++i) {
        int e = i * 256 + t;
        int n = e & 63, kk = e >> 6;
        int gn = n0 + n, gk = k0 + kk;
        int nc = gn < N ? gn : N - 1;
        int kc = gk < K ? gk : K - 1;
        float msk = (gn < N && gk < K) ? 1.f : 0.f;
        Bs[n][kk] = B[(size_t)kc * ldb + nc] * msk;
      }
    }
    __syncthreads();
#pragma unroll
    for (int ks = 0; ks < 4; ++ks) {
      int kb = ks * 4 + kh;
      v2f a;
      a.x = As[wm * 16 + lm][kb];
      a.y = As[wm * 16 + lm][kb + 1];
#pragma unroll
      for (int nt = 0; nt < 2; ++nt) {
        int n = wn * 32 + nt * 16 + lm;
        v2f bb;
        bb.x = Bs[n][kb];
        bb.y = Bs[n][kb + 1];
        acc[nt] = __builtin_amdgcn_wmma_f32_16x16x4_f32(
            false, a, false, bb, (short)0, acc[nt], false, false);
      }
    }
    __syncthreads();
  }
#pragma unroll
  for (int nt = 0; nt < 2; ++nt) {
#pragma unroll
    for (int g = 0; g < 8; ++g) {
      int gm = m0 + wm * 16 + ((lane < 16) ? g : g + 8);
      int gn = n0 + wn * 32 + nt * 16 + lm;
      if (gm < M && gn < N) C[(size_t)gm * ldc + gn] = alpha * acc[nt][g];
    }
  }
}

// ---------------------------------------------------------------------------
// Seed the H chain: HT[t, q] = (Ad^t Bd)[q] for t = 0..15 (sequential GEMVs).
// ---------------------------------------------------------------------------
__global__ void k_seed_H(const float* __restrict__ E, float* __restrict__ HT) {
  const int t = threadIdx.x;   // 768 threads
  __shared__ float h[768];
  h[t] = E[(size_t)t * PADN + Dm];
  __syncthreads();
  HT[t] = h[t];
  for (int step = 1; step < 16; ++step) {
    float s = 0.f;
    const float* row = E + (size_t)t * PADN;
    for (int c = 0; c < Dm; ++c) s += row[c] * h[c];
    __syncthreads();
    h[t] = s;
    __syncthreads();
    HT[(size_t)step * Dm + t] = s;
  }
}

// ---------------------------------------------------------------------------
// u[b,l] = relu(sum_d x[b,d,l]*W_u[d] + b_u), and xT_bf16[b,l,d] = bf16(x[b,d,l])
// ---------------------------------------------------------------------------
__global__ __launch_bounds__(256) void k_u_xT(
    const float* __restrict__ x, const float* __restrict__ Wu,
    const float* __restrict__ bu, float* __restrict__ u,
    unsigned short* __restrict__ xT) {
  __shared__ float tile[64][129];
  __shared__ float wsh[64];
  __shared__ float ured[256];
  const int b = blockIdx.x;
  const int t = threadIdx.x;
  const int l = t & 127, dg = t >> 7;           // dg in {0,1}
  float acc = 0.f;
  const float* xb = x + (size_t)b * Dm * Lseq;
  unsigned short* xtb = xT + (size_t)b * Lseq * Dm;
  const int dd = t & 63, l2base = t >> 6;       // write mapping
  for (int d0 = 0; d0 < Dm; d0 += 64) {
    if (t < 64) wsh[t] = Wu[d0 + t];
    __syncthreads();
#pragma unroll 4
    for (int i = 0; i < 32; ++i) {
      int dl = dg * 32 + i;
      float v = xb[(size_t)(d0 + dl) * Lseq + l];
      tile[dl][l] = v;
      acc += v * wsh[dl];
    }
    __syncthreads();
#pragma unroll 4
    for (int i = 0; i < 32; ++i) {
      int l2 = l2base + i * 4;
      xtb[(size_t)l2 * Dm + d0 + dd] = f2bf(tile[dd][l2]);
    }
    __syncthreads();
  }
  ured[t] = acc;
  __syncthreads();
  if (t < 128) {
    float s = ured[t] + ured[t + 128] + bu[0];
    u[(size_t)b * Lseq + t] = fmaxf(s, 0.f);
  }
}

// ---------------------------------------------------------------------------
// GEMM 2 (bf16 WMMA): h[b*128+l, o] = relu( T_b @ G  +  xT_b @ W_h2^T + b_h )
// K-steps 0..3: A = Toeplitz(u_b) generated in LDS, B = G (fp32->bf16).
// K-steps 4..27: A tile DMA'd by TDM from xT, B = W_h (fp32->bf16, transposed).
// ---------------------------------------------------------------------------
__global__ __launch_bounds__(256) void k_gemm_h(
    const float* __restrict__ u, const unsigned short* __restrict__ xT,
    const float* __restrict__ G32, const float* __restrict__ Wh,
    const float* __restrict__ bh, unsigned short* __restrict__ hout) {
  __shared__ __align__(16) unsigned short Ash[128][40];
  __shared__ __align__(16) unsigned short Bsh[128][40];   // Bsh[n][k]
  __shared__ float ush[128];
  const int t = threadIdx.x;
  const int b = blockIdx.y;
  const int n0 = blockIdx.x * 128;
  if (t < 128) ush[t] = u[(size_t)b * Lseq + t];
  const int wave = t >> 5, lane = t & 31;
  const int wm = wave >> 2, wn = wave & 3;
  const int lm = lane & 15;
  const int kbA = (lane >> 4) << 4;   // A frag: K base 0 or 16 (contiguous 16)
  const int kbB = (lane >> 4) << 4;   // B frag: K base 0 or 16 (contiguous 16)
  const unsigned ashBase = (unsigned)(uintptr_t)&Ash[0][0];
  const unsigned short* xA = xT + (size_t)b * Lseq * Dm;
  v8f acc[4][2];
#pragma unroll
  for (int i = 0; i < 4; ++i)
#pragma unroll
    for (int j = 0; j < 2; ++j)
#pragma unroll
      for (int e = 0; e < 8; ++e) acc[i][j][e] = 0.f;
  __syncthreads();

  for (int step = 0; step < 28; ++step) {
    const int kg0 = step * 32;
    if (step < 4) {
#pragma unroll
      for (int i = 0; i < 16; ++i) {               // Toeplitz A[l,k] = u[l-k]
        int e = i * 256 + t;
        int l = e >> 5, kk = e & 31;
        int kg = kg0 + kk;
        Ash[l][kk] = (kg <= l) ? f2bf(ush[l - kg]) : (unsigned short)0;
      }
#pragma unroll
      for (int i = 0; i < 16; ++i) {               // B = G32[k, n0+n]
        int e = i * 256 + t;
        int n = e & 127, kk = e >> 7;
        Bsh[n][kk] = f2bf(G32[(size_t)(kg0 + kk) * Dm + n0 + n]);
      }
    } else {
#if HAVE_TDM
      if (wave == 0)                               // A tile: TDM DMA, padded
        tdm_load_2d(xA + (kg0 - 128), ashBase, 16, 128, 384);
#else
      const unsigned short* srcA = xA + (kg0 - 128);
#pragma unroll
      for (int i = 0; i < 16; ++i) {
        int e = i * 256 + t;
        int l = e >> 5, kk = e & 31;
        Ash[l][kk] = srcA[(size_t)l * Dm + kk];
      }
#endif
      const float* srcB = Wh + (size_t)(640 + kg0);  // col = 768 + (kg0-128)
#pragma unroll
      for (int i = 0; i < 16; ++i) {               // B[k,n] = Wh[n, 768+k]
        int e = i * 256 + t;
        int kk = e & 31, n = e >> 5;
        Bsh[n][kk] = f2bf(srcB[(size_t)(n0 + n) * 1536 + kk]);
      }
      if (step < 27)                               // prefetch next B panel
        __builtin_prefetch(srcB + 32 + (size_t)(n0 + (t >> 3)) * 1536, 0, 1);
    }
#if HAVE_TDM
    __builtin_amdgcn_s_wait_tensorcnt(0);
#endif
    __syncthreads();
    FragB16 fb[2];
#pragma unroll
    for (int nt = 0; nt < 2; ++nt) {
      int n = wn * 32 + nt * 16 + lm;
      fb[nt].q[0] = *(const uint4*)&Bsh[n][kbB];
      fb[nt].q[1] = *(const uint4*)&Bsh[n][kbB + 8];
    }
#pragma unroll
    for (int mt = 0; mt < 4; ++mt) {
      int m = wm * 64 + mt * 16 + lm;
      FragB16 fa;
      fa.q[0] = *(const uint4*)&Ash[m][kbA];
      fa.q[1] = *(const uint4*)&Ash[m][kbA + 8];
#pragma unroll
      for (int nt = 0; nt < 2; ++nt)
        acc[mt][nt] = __builtin_amdgcn_wmma_f32_16x16x32_bf16(
            false, fa.v, false, fb[nt].v, (short)0, acc[mt][nt], false, false);
    }
    __syncthreads();
  }
  // epilogue: + b_h, relu, bf16 store
  unsigned short* hb = hout + (size_t)b * Lseq * Dm + n0;
#pragma unroll
  for (int nt = 0; nt < 2; ++nt) {
    int gn = wn * 32 + nt * 16 + lm;
    float bias = bh[n0 + gn];
#pragma unroll
    for (int mt = 0; mt < 4; ++mt) {
#pragma unroll
      for (int g = 0; g < 8; ++g) {
        int gm = wm * 64 + mt * 16 + ((lane < 16) ? g : g + 8);
        float v = acc[mt][nt][g] + bias;
        hb[(size_t)gm * Dm + gn] = f2bf(fmaxf(v, 0.f));
      }
    }
  }
}

// ---------------------------------------------------------------------------
// GEMM 3 (bf16 WMMA): y[b,o2,l] = BN( h[b,l,:] @ conv_w[o2,:] + conv_b[o2] )
// A tile via TDM; epilogue does BN affine and transposes through LDS so the
// 100 MB output store is fully coalesced along l.
// ---------------------------------------------------------------------------
__global__ __launch_bounds__(256) void k_gemm_y(
    const unsigned short* __restrict__ hbuf, const float* __restrict__ cw,
    const float* __restrict__ cb, const float* __restrict__ gam,
    const float* __restrict__ bet, const float* __restrict__ mea,
    const float* __restrict__ var, float* __restrict__ out) {
  __shared__ __align__(16) unsigned short Ash[128][40];
  __shared__ __align__(16) unsigned short Bsh[128][40];
  __shared__ float ysh[128][133];
  const int t = threadIdx.x;
  const int b = blockIdx.y;
  const int n0 = blockIdx.x * 128;
  const int wave = t >> 5, lane = t & 31;
  const int wm = wave >> 2, wn = wave & 3;
  const int lm = lane & 15;
  const int kbA = (lane >> 4) << 4;
  const int kbB = (lane >> 4) << 4;
  const unsigned ashBase = (unsigned)(uintptr_t)&Ash[0][0];
  v8f acc[4][2];
#pragma unroll
  for (int i = 0; i < 4; ++i)
#pragma unroll
    for (int j = 0; j < 2; ++j)
#pragma unroll
      for (int e = 0; e < 8; ++e) acc[i][j][e] = 0.f;

  const unsigned short* hA = hbuf + (size_t)b * Lseq * Dm;
  for (int step = 0; step < 24; ++step) {
    const int kg0 = step * 32;
#if HAVE_TDM
    if (wave == 0)
      tdm_load_2d(hA + kg0, ashBase, 16, 128, 384);
#else
#pragma unroll
    for (int i = 0; i < 16; ++i) {
      int e = i * 256 + t;
      int l = e >> 5, kk = e & 31;
      Ash[l][kk] = hA[(size_t)l * Dm + kg0 + kk];
    }
#endif
#pragma unroll
    for (int i = 0; i < 16; ++i) {                 // B[k,n] = cw[n0+n, k]
      int e = i * 256 + t;
      int kk = e & 31, n = e >> 5;
      Bsh[n][kk] = f2bf(cw[(size_t)(n0 + n) * Dm + kg0 + kk]);
    }
    if (step < 23)
      __builtin_prefetch(cw + (size_t)(n0 + (t >> 3)) * Dm + kg0 + 32, 0, 1);
#if HAVE_TDM
    __builtin_amdgcn_s_wait_tensorcnt(0);
#endif
    __syncthreads();
    FragB16 fb[2];
#pragma unroll
    for (int nt = 0; nt < 2; ++nt) {
      int n = wn * 32 + nt * 16 + lm;
      fb[nt].q[0] = *(const uint4*)&Bsh[n][kbB];
      fb[nt].q[1] = *(const uint4*)&Bsh[n][kbB + 8];
    }
#pragma unroll
    for (int mt = 0; mt < 4; ++mt) {
      int m = wm * 64 + mt * 16 + lm;
      FragB16 fa;
      fa.q[0] = *(const uint4*)&Ash[m][kbA];
      fa.q[1] = *(const uint4*)&Ash[m][kbA + 8];
#pragma unroll
      for (int nt = 0; nt < 2; ++nt)
        acc[mt][nt] = __builtin_amdgcn_wmma_f32_16x16x32_bf16(
            false, fa.v, false, fb[nt].v, (short)0, acc[mt][nt], false, false);
    }
    __syncthreads();
  }
  // epilogue: conv_b + batchnorm affine into LDS tile (row = l, col = o2)
#pragma unroll
  for (int nt = 0; nt < 2; ++nt) {
    int gn = wn * 32 + nt * 16 + lm;
    int o2 = n0 + gn;
    float invv = gam[o2] * rsqrtf(var[o2] + 1e-5f);
    float off = (cb[o2] - mea[o2]) * invv + bet[o2];
#pragma unroll
    for (int mt = 0; mt < 4; ++mt) {
#pragma unroll
      for (int g = 0; g < 8; ++g) {
        int gm = wm * 64 + mt * 16 + ((lane < 16) ? g : g + 8);
        ysh[gm][gn] = acc[mt][nt][g] * invv + off;
      }
    }
  }
  __syncthreads();
  float* ob = out + ((size_t)b * Dm + n0) * Lseq;
  const int l = t & 127;
#pragma unroll 4
  for (int i = 0; i < 64; ++i) {
    int oc = i * 2 + (t >> 7);
    ob[(size_t)oc * Lseq + l] = ysh[l][oc];
  }
}

// ---------------------------------------------------------------------------
static inline void gemm_f32(const float* A, int lda, const float* B, int ldb,
                            float* C, int ldc, int M, int N, int K, float alpha,
                            int transB, hipStream_t s) {
  dim3 g((N + 63) / 64, (M + 63) / 64);
  k_gemm_f32<<<g, 256, 0, s>>>(A, lda, B, ldb, C, ldc, M, N, K, alpha, transB);
}

extern "C" void kernel_launch(void* const* d_in, const int* in_sizes, int n_in,
                              void* d_out, int out_size, void* d_ws, size_t ws_size,
                              hipStream_t stream) {
  (void)in_sizes; (void)n_in; (void)out_size; (void)ws_size;
  const float* x    = (const float*)d_in[0];
  const float* Wu   = (const float*)d_in[1];
  const float* bu   = (const float*)d_in[2];
  const float* Wh   = (const float*)d_in[3];
  const float* bh   = (const float*)d_in[4];
  const float* cw   = (const float*)d_in[5];
  const float* cb   = (const float*)d_in[6];
  const float* gam  = (const float*)d_in[7];
  const float* bet  = (const float*)d_in[8];
  const float* mea  = (const float*)d_in[9];
  const float* var  = (const float*)d_in[10];
  float* out = (float*)d_out;

  char* ws = (char*)d_ws;
  const size_t SQ = (size_t)NSQ * 4;            // 2.46 MB per 784x784 buffer
  float* X  = (float*)(ws);
  float* E  = (float*)(ws + SQ);
  float* E2 = (float*)(ws + 2 * SQ);
  float* P  = (float*)(ws + 3 * SQ);
  float* P2 = (float*)(ws + 4 * SQ);
  float* u  = (float*)(ws + 5 * SQ);
  unsigned short* xT   = (unsigned short*)(ws + 5 * SQ + (size_t)131072);
  unsigned short* hbuf = (unsigned short*)(ws + 5 * SQ + (size_t)131072 +
                                           (size_t)Bsz * Lseq * Dm * 2);
  // overlays (live only after the Taylor stage is done):
  float* QA  = X;                                   // Ad^2, Ad^8
  float* QB  = P;                                   // Ad^4, Ad^16
  float* HT  = P2;                                  // 128 x 768 fp32
  float* G32 = (float*)((char*)P2 + 524288);        // 128 x 768 fp32

  // ---- input-dependent prep (independent of expm chain) ----
  k_u_xT<<<Bsz, 256, 0, stream>>>(x, Wu, bu, u, xT);

  // ---- expm(Maug) via scaling (2^-14) + Taylor(12) + 14 squarings ----
  const int gb = (NSQ + 255) / 256;
  k_build_X<<<gb, 256, 0, stream>>>(X);
  k_init_EP<<<gb, 256, 0, stream>>>(X, E, P);
  float* pc = P; float* pn = P2;
  for (int k = 2; k <= TAYLOR_K; ++k) {
    gemm_f32(pc, PADN, X, PADN, pn, PADN, PADN, PADN, PADN, 1.0f / (float)k, 0, stream);
    k_axpy<<<gb, 256, 0, stream>>>(E, pn, NSQ);
    float* tsw = pc; pc = pn; pn = tsw;
  }
  float* ec = E; float* ea = E2;
  for (int i = 0; i < EXP_S; ++i) {
    gemm_f32(ec, PADN, ec, PADN, ea, PADN, PADN, PADN, PADN, 1.0f, 0, stream);
    float* tsw = ec; ec = ea; ea = tsw;
  }
  // ec now holds expm(Maug): Ad = ec[0:768,0:768] (ld PADN), Bd = ec[:,768]

  // ---- H chain: HT[t,:] = Ad^t Bd ----
  k_seed_H<<<1, Dm, 0, stream>>>(ec, HT);                      // rows 0..15
  gemm_f32(ec, PADN, ec, PADN, QA, Dm, Dm, Dm, Dm, 1.0f, 0, stream);  // Ad^2
  gemm_f32(QA, Dm, QA, Dm, QB, Dm, Dm, Dm, Dm, 1.0f, 0, stream);      // Ad^4
  gemm_f32(QB, Dm, QB, Dm, QA, Dm, Dm, Dm, Dm, 1.0f, 0, stream);      // Ad^8
  gemm_f32(QA, Dm, QA, Dm, QB, Dm, Dm, Dm, Dm, 1.0f, 0, stream);      // Ad^16
  for (int g = 1; g < 8; ++g)   // rows 16g..16g+15 = (prev block) @ (Ad^16)^T
    gemm_f32(HT + (size_t)(g - 1) * 16 * Dm, Dm, QB, Dm,
             HT + (size_t)g * 16 * Dm, Dm, 16, Dm, Dm, 1.0f, 1, stream);

  // ---- fold impulse response into layer: G = HT @ W_h[:, :768]^T ----
  gemm_f32(HT, Dm, Wh, 2 * Dm, G32, Dm, Lseq, Dm, Dm, 1.0f, 1, stream);

  // ---- main bf16 WMMA GEMMs ----
  k_gemm_h<<<dim3(Dm / 128, Bsz), 256, 0, stream>>>(u, xT, G32, Wh, bh, hbuf);
  k_gemm_y<<<dim3(Dm / 128, Bsz), 256, 0, stream>>>(hbuf, cw, cb, gam, bet, mea,
                                                    var, out);
}
